// LearnableNetwork_10256381903546
// MI455X (gfx1250) — compile-verified, matrix operand order checked by reading, not run on hardware
//
#include <hip/hip_runtime.h>
#include <hip/hip_bf16.h>

typedef __bf16 bf16_t;
typedef __attribute__((ext_vector_type(16))) __bf16 v16bf;
typedef __attribute__((ext_vector_type(8)))  __bf16 v8bf;
typedef __attribute__((ext_vector_type(8)))  float  v8f;

#define B_    32
#define NOPS_ 1000
#define J_    100
#define M_    50
#define D_    128
#define PAIR_ 8
#define HA_   512
#define HC_   256
#define NROWS (J_*M_)            /* 5000 rows per batch */
#define KIN   520
#define KINP  544                /* pad to 17 * 32 */
#define RPB   64                 /* rows per block */
#define TILES64 ((NROWS + RPB - 1) / RPB)   /* 79 */

union VAcc { v8f v; float f[8]; };
union Frag { v16bf v; v8bf h[2]; };

// CDNA5 has a hardware transcendental tanh (V_TANH_F32); use it if the
// toolchain exposes the builtin, else fall back to the precise libm expansion.
__device__ __forceinline__ float fast_tanh(float x) {
#if __has_builtin(__builtin_amdgcn_tanhf)
    return __builtin_amdgcn_tanhf(x);
#else
    return tanhf(x);
#endif
}

// ---------------------------------------------------------------------------
// Weight prep: aw1 [520x512] f32 (KxN row major) -> W1t bf16 [512][544] (NxK,
// zero padded).  aw2 [512x512] -> W2t bf16 [512][512] (NxK).
// ---------------------------------------------------------------------------
__global__ __launch_bounds__(256)
void convert_weights_kernel(const float* __restrict__ aw1,
                            const float* __restrict__ aw2,
                            bf16_t* __restrict__ W1t,
                            bf16_t* __restrict__ W2t) {
    int idx = blockIdx.x * 256 + threadIdx.x;
    if (idx < HA_ * KINP) {
        int n = idx / KINP, k = idx % KINP;
        float v = (k < KIN) ? aw1[k * HA_ + n] : 0.0f;
        W1t[idx] = (bf16_t)v;
    }
    if (idx < HA_ * HA_) {
        int n = idx / HA_, k = idx % HA_;
        W2t[idx] = (bf16_t)aw2[k * HA_ + n];
    }
}

// ---------------------------------------------------------------------------
// One dense layer over a 64-row tile:
//   Hout[64][512] = tanh( X[64][KT*32] @ Wt^T + bias )
// 8 waves: each wave owns 4 N-tiles (64 cols) and all 4 M-tiles -> 16
// accumulators, so each B fragment (streamed from L2-resident transposed
// weights) feeds 4 WMMAs and 16 independent chains hide load latency.
// Fragment layout per ISA 7.12.2 (16-bit 16x32): lanes 0-15 hold
// K in {kb..kb+7, kb+16..kb+23}, lanes 16-31 hold {kb+8.., kb+24..}.
// ---------------------------------------------------------------------------
__device__ __forceinline__
void mlp_layer_wmma64(const bf16_t* __restrict__ Xs, int ldx, int KT,
                      const bf16_t* __restrict__ Wt, int ldk,
                      const float* __restrict__ bias,
                      bf16_t* __restrict__ Hout) {
    const int lane  = threadIdx.x & 31;
    const int wv    = threadIdx.x >> 5;
    const int nlo   = lane & 15;
    const int khalf = (lane & 16) ? 8 : 0;
    const int r0    = (lane & 16) ? 8 : 0;

    VAcc acc[4][4];
    #pragma unroll
    for (int mt = 0; mt < 4; ++mt)
        #pragma unroll
        for (int nt = 0; nt < 4; ++nt)
            #pragma unroll
            for (int i = 0; i < 8; ++i) acc[mt][nt].f[i] = 0.0f;

    for (int kt = 0; kt < KT; ++kt) {
        const int kb = kt * 32 + khalf;
        Frag a[4];
        #pragma unroll
        for (int mt = 0; mt < 4; ++mt) {
            const bf16_t* pa = Xs + (size_t)(mt * 16 + nlo) * ldx + kb;
            a[mt].h[0] = *(const v8bf*)(pa);
            a[mt].h[1] = *(const v8bf*)(pa + 16);
        }
        #pragma unroll
        for (int nt = 0; nt < 4; ++nt) {
            const int ncol = (wv * 4 + nt) * 16 + nlo;
            Frag bfr;
            const bf16_t* pb = Wt + (size_t)ncol * ldk + kb;
            bfr.h[0] = *(const v8bf*)(pb);
            bfr.h[1] = *(const v8bf*)(pb + 16);
            #pragma unroll
            for (int mt = 0; mt < 4; ++mt) {
                acc[mt][nt].v = __builtin_amdgcn_wmma_f32_16x16x32_bf16(
                    false, a[mt].v, false, bfr.v,
                    (short)0, acc[mt][nt].v, false, false);
            }
        }
    }

    // Epilogue: bias + tanh, store bf16 activations for the next layer.
    // C/D layout: lane<16 -> rows 0..7 (vgpr i = row i), lane>=16 -> rows 8..15.
    #pragma unroll
    for (int nt = 0; nt < 4; ++nt) {
        const int ncol = (wv * 4 + nt) * 16 + nlo;
        const float bz = bias[ncol];
        #pragma unroll
        for (int mt = 0; mt < 4; ++mt) {
            #pragma unroll
            for (int i = 0; i < 8; ++i) {
                float tv = fast_tanh(acc[mt][nt].f[i] + bz);
                Hout[(size_t)(mt * 16 + r0 + i) * HA_ + ncol] = (bf16_t)tv;
            }
        }
    }
}

// ---------------------------------------------------------------------------
// Actor scores: one block per (64-row tile, batch).
// ---------------------------------------------------------------------------
__global__ __launch_bounds__(256)
void actor_scores_kernel(const float* __restrict__ fea_j,
                         const float* __restrict__ fea_m,
                         const float* __restrict__ fea_jg,
                         const float* __restrict__ fea_mg,
                         const float* __restrict__ fea_pairs,
                         const int* __restrict__ candidate,
                         const unsigned char* __restrict__ mask,
                         const bf16_t* __restrict__ W1t,
                         const float* __restrict__ ab1,
                         const bf16_t* __restrict__ W2t,
                         const float* __restrict__ ab2,
                         const float* __restrict__ aw3,
                         const float* __restrict__ ab3,
                         float* __restrict__ scores) {
    // bufA: X [64][544] for layer 1, reused as H2 [64][512] for layer 2 out.
    // bufB: H1 [64][512].
    __shared__ __align__(16) bf16_t bufA[RPB * KINP];
    __shared__ __align__(16) bf16_t bufB[RPB * HA_];
    __shared__ float w3s[HA_];
    __shared__ float part[256];

    const int b = blockIdx.y;
    const int rowbase = blockIdx.x * RPB;
    const int t = threadIdx.x;

    // Stage layer-3 weights.
    for (int k = t; k < HA_; k += 256) w3s[k] = aw3[k];

    // Stage concatenated features [fj | fm | gj | gm | pairs | 0-pad] as bf16.
    for (int idx = t; idx < RPB * KINP; idx += 256) {
        const int i = idx / KINP, k = idx % KINP;
        const int r = rowbase + i;
        float val = 0.0f;
        if (r < NROWS && k < KIN) {
            const int j = r / M_, m = r - j * M_;
            if (k < 128) {
                const int cj = candidate[b * J_ + j];
                val = fea_j[((size_t)b * NOPS_ + cj) * D_ + k];
            } else if (k < 256) {
                val = fea_m[((size_t)b * M_ + m) * D_ + (k - 128)];
            } else if (k < 384) {
                val = fea_jg[b * D_ + (k - 256)];
            } else if (k < 512) {
                val = fea_mg[b * D_ + (k - 384)];
            } else {
                val = fea_pairs[(((size_t)b * J_ + j) * M_ + m) * PAIR_ + (k - 512)];
            }
        }
        bufA[idx] = (bf16_t)val;
    }
    __syncthreads();

    mlp_layer_wmma64(bufA, KINP, KINP / 32, W1t, KINP, ab1, bufB);  // 520->512
    __syncthreads();
    mlp_layer_wmma64(bufB, HA_, HA_ / 32, W2t, HA_, ab2, bufA);     // 512->512
    __syncthreads();

    // Layer 3: 512 -> 1 per row (H2 now lives in bufA with ld=512).
    // 4 threads per row, 128-element partial dots, then combine + mask.
    {
        const int r = t >> 2;          // 0..63
        const int q = t & 3;           // quarter of the k range
        const bf16_t* hrow = bufA + (size_t)r * HA_ + q * 128;
        const float*  wq   = w3s + q * 128;
        float s = 0.0f;
        for (int k = 0; k < 128; ++k) s += (float)hrow[k] * wq[k];
        part[t] = s;
    }
    __syncthreads();
    if (t < RPB) {
        const int r = rowbase + t;
        if (r < NROWS) {
            float s = part[t * 4] + part[t * 4 + 1]
                    + part[t * 4 + 2] + part[t * 4 + 3] + ab3[0];
            const int j = r / M_, m = r - j * M_;
            const bool msk = mask[((size_t)b * J_ + j) * M_ + m] != 0;
            scores[(size_t)b * NROWS + r] = msk ? -INFINITY : s;
        }
    }
}

// ---------------------------------------------------------------------------
// Softmax over 5000 scores per batch (masked entries are -inf -> exp == 0).
// ---------------------------------------------------------------------------
__global__ __launch_bounds__(256)
void softmax_kernel(const float* __restrict__ scores, float* __restrict__ pi) {
    __shared__ float red[256];
    const int b = blockIdx.x, t = threadIdx.x;
    const float* s = scores + (size_t)b * NROWS;

    float mx = -INFINITY;
    for (int i = t; i < NROWS; i += 256) mx = fmaxf(mx, s[i]);
    red[t] = mx; __syncthreads();
    for (int off = 128; off > 0; off >>= 1) {
        if (t < off) red[t] = fmaxf(red[t], red[t + off]);
        __syncthreads();
    }
    mx = red[0]; __syncthreads();

    float sum = 0.0f;
    for (int i = t; i < NROWS; i += 256) sum += __expf(s[i] - mx);
    red[t] = sum; __syncthreads();
    for (int off = 128; off > 0; off >>= 1) {
        if (t < off) red[t] += red[t + off];
        __syncthreads();
    }
    const float inv = 1.0f / red[0];
    for (int i = t; i < NROWS; i += 256)
        pi[(size_t)b * NROWS + i] = __expf(s[i] - mx) * inv;
}

// ---------------------------------------------------------------------------
// Critic: 32 rows of 256->256->256->1 tanh MLP, pure f32 VALU (negligible).
// ---------------------------------------------------------------------------
__global__ __launch_bounds__(256)
void critic_kernel(const float* __restrict__ fjg, const float* __restrict__ fmg,
                   const float* __restrict__ cw1, const float* __restrict__ cb1,
                   const float* __restrict__ cw2, const float* __restrict__ cb2,
                   const float* __restrict__ cw3, const float* __restrict__ cb3,
                   float* __restrict__ vout) {
    __shared__ float g[2 * D_];
    __shared__ float h1[HC_];
    __shared__ float h2[HC_];
    __shared__ float red[HC_];
    const int b = blockIdx.x, t = threadIdx.x;
    if (t < D_) { g[t] = fjg[b * D_ + t]; g[D_ + t] = fmg[b * D_ + t]; }
    __syncthreads();
    {
        float s = cb1[t];
        for (int k = 0; k < 2 * D_; ++k) s += g[k] * cw1[k * HC_ + t];
        h1[t] = fast_tanh(s);
    }
    __syncthreads();
    {
        float s = cb2[t];
        for (int k = 0; k < HC_; ++k) s += h1[k] * cw2[k * HC_ + t];
        h2[t] = fast_tanh(s);
    }
    __syncthreads();
    red[t] = h2[t] * cw3[t];
    __syncthreads();
    for (int off = 128; off > 0; off >>= 1) {
        if (t < off) red[t] += red[t + off];
        __syncthreads();
    }
    if (t == 0) vout[b] = red[0] + cb3[0];
}

// ---------------------------------------------------------------------------
extern "C" void kernel_launch(void* const* d_in, const int* in_sizes, int n_in,
                              void* d_out, int out_size, void* d_ws, size_t ws_size,
                              hipStream_t stream) {
    const float* fea_j      = (const float*)d_in[0];
    const float* fea_m      = (const float*)d_in[1];
    const float* fea_jg     = (const float*)d_in[2];
    const float* fea_mg     = (const float*)d_in[3];
    const float* fea_pairs  = (const float*)d_in[4];
    const int*   candidate  = (const int*)d_in[5];
    const unsigned char* mask = (const unsigned char*)d_in[6];
    const float* aw1 = (const float*)d_in[7];
    const float* ab1 = (const float*)d_in[8];
    const float* aw2 = (const float*)d_in[9];
    const float* ab2 = (const float*)d_in[10];
    const float* aw3 = (const float*)d_in[11];
    const float* ab3 = (const float*)d_in[12];
    const float* cw1 = (const float*)d_in[13];
    const float* cb1 = (const float*)d_in[14];
    const float* cw2 = (const float*)d_in[15];
    const float* cb2 = (const float*)d_in[16];
    const float* cw3 = (const float*)d_in[17];
    const float* cb3 = (const float*)d_in[18];

    // Workspace: W1t bf16 [512][544] | W2t bf16 [512][512] | scores f32 [32][5000]
    char* ws = (char*)d_ws;
    bf16_t* W1t    = (bf16_t*)(ws);
    bf16_t* W2t    = (bf16_t*)(ws + (size_t)HA_ * KINP * sizeof(bf16_t));
    float*  scores = (float*)(ws + (size_t)HA_ * KINP * sizeof(bf16_t)
                                  + (size_t)HA_ * HA_ * sizeof(bf16_t));

    float* pi   = (float*)d_out;                  // [32, 5000]
    float* vout = pi + (size_t)B_ * NROWS;        // [32, 1]

    convert_weights_kernel<<<(HA_ * KINP + 255) / 256, 256, 0, stream>>>(
        aw1, aw2, W1t, W2t);

    dim3 grid(TILES64, B_);
    actor_scores_kernel<<<grid, 256, 0, stream>>>(
        fea_j, fea_m, fea_jg, fea_mg, fea_pairs, candidate, mask,
        W1t, ab1, W2t, ab2, aw3, ab3, scores);

    softmax_kernel<<<B_, 256, 0, stream>>>(scores, pi);

    critic_kernel<<<B_, 256, 0, stream>>>(
        fea_jg, fea_mg, cw1, cb1, cw2, cb2, cw3, cb3, vout);
}
